// VaeCritic_42984032698404
// MI455X (gfx1250) — compile-verified, matrix operand order checked by reading, not run on hardware
//
#include <hip/hip_runtime.h>
#include <hip/hip_bf16.h>

// ---------------------------------------------------------------------------
// MoE critic for MI455X (gfx1250): bf16 WMMA GEMM chain.
//   B=8192, D=512, V=64, E=8, HIDDEN=(1024,512,256)
// ~167 GFLOP vs ~60 MB minimal traffic -> heavily compute-bound; run all
// layers through v_wmma_f32_16x16x32_bf16 (f32 accumulate).
// GEMM: 128x128 block tile, 8 waves x (64x32) wave tiles, BK=32,
// LDS double-buffered (1 barrier / K-step), LDS-staged coalesced b128 stores.
// ---------------------------------------------------------------------------

typedef __attribute__((ext_vector_type(16))) __bf16 bf16x16;
typedef __attribute__((ext_vector_type(8)))  __bf16 bf16x8;
typedef __attribute__((ext_vector_type(8)))  float  f32x8;

#define NB   8192
#define ND   512
#define NV   64
#define NE   8

#define BM   128
#define BN   128
#define BK   32
#define LDSS 40    // padded LDS row stride (elements): 80B rows, conflict-free b128
#define CSTR 136   // epilogue staging stride (elements): 272B rows, 16B aligned

// ------------------------- cast / transpose kernels -------------------------

__global__ void cast_f32_to_bf16(const float* __restrict__ src,
                                 __bf16* __restrict__ dst, int n) {
  int i = blockIdx.x * blockDim.x + threadIdx.x;
  if (i < n) dst[i] = (__bf16)src[i];
}

// src: [E, di, do] f32   ->   dst: [E, do, di] bf16  (per-expert transpose)
__global__ void transpose_cast_bf16(const float* __restrict__ src,
                                    __bf16* __restrict__ dst,
                                    int E, int di, int dof) {
  int idx = blockIdx.x * blockDim.x + threadIdx.x;
  int total = E * di * dof;
  if (idx >= total) return;
  int e   = idx / (di * dof);
  int rem = idx % (di * dof);
  int o   = rem / di;
  int i   = rem % di;
  dst[idx] = (__bf16)src[(long)e * di * dof + (long)i * dof + o];
}

// ------------------------------ WMMA GEMM -----------------------------------
// C[e] = act(A[e][M,K] * W[e][N,K]^T + bias[e][N]) ; A row-major (lda=K-ish),
// W stored transposed row-major [N,K], C row-major (ldc=N), bf16 in/out,
// f32 accumulate. grid = (N/BN, M/BM, E), block = 256 (8 waves).

__global__ __launch_bounds__(256)
void gemm_bf16_wmma(const __bf16* __restrict__ A, long strideA, int lda,
                    const __bf16* __restrict__ W, long strideW,
                    const float*  __restrict__ bias, int strideBias,
                    __bf16* __restrict__ C, long strideC, int ldc,
                    int K, int doElu) {
  // As[2] | Bs[2] double buffers; epilogue reuses the whole region as Cs.
  __shared__ __bf16 smem[2 * BM * LDSS + 2 * BN * LDSS];  // 40 KB
  __bf16* const AsBase = smem;
  __bf16* const BsBase = smem + 2 * BM * LDSS;

  const int e = blockIdx.z;
  A    += (long)e * strideA;
  W    += (long)e * strideW;
  bias += (long)e * strideBias;
  C    += (long)e * strideC;

  const int m0   = blockIdx.y * BM;
  const int n0   = blockIdx.x * BN;
  const int tid  = threadIdx.x;
  const int lane = tid & 31;
  const int wave = tid >> 5;
  const int wm   = (wave & 1) * 64;   // wave row offset inside block tile
  const int wn   = (wave >> 1) * 32;  // wave col offset inside block tile

  // Global->reg staging: 512 chunks of 16B per tile side, 2 chunks/thread.
  bf16x8 ra[2], rb[2];
  const int kTiles = K / BK;

  auto loadRegs = [&](int kt) {
    const int k0 = kt * BK;
#pragma unroll
    for (int j = 0; j < 2; ++j) {
      int c  = tid + j * 256;
      int r  = c >> 2;          // tile row (0..127)
      int kc = (c & 3) * 8;     // k-chunk offset (elements)
      ra[j] = *(const bf16x8*)(A + (long)(m0 + r) * lda + k0 + kc);
      rb[j] = *(const bf16x8*)(W + (long)(n0 + r) * K   + k0 + kc);
    }
  };
  auto storeLds = [&](int s) {
    __bf16* As = AsBase + s * (BM * LDSS);
    __bf16* Bs = BsBase + s * (BN * LDSS);
#pragma unroll
    for (int j = 0; j < 2; ++j) {
      int c  = tid + j * 256;
      int r  = c >> 2;
      int kc = (c & 3) * 8;
      *(bf16x8*)&As[r * LDSS + kc] = ra[j];
      *(bf16x8*)&Bs[r * LDSS + kc] = rb[j];
    }
  };

  f32x8 acc[4][2];
#pragma unroll
  for (int mi = 0; mi < 4; ++mi)
#pragma unroll
    for (int ni = 0; ni < 2; ++ni)
      acc[mi][ni] = (f32x8){0.f, 0.f, 0.f, 0.f, 0.f, 0.f, 0.f, 0.f};

  // WMMA 16-bit fragment addressing (ISA 7.12.2):
  //  lanes 0-15: row m16, K 0-7 (v0-3) & 16-23 (v4-7); lanes 16-31: K 8-15 & 24-31.
  const int m16 = lane & 15;
  const int kh  = (lane >> 4) * 8;

  loadRegs(0);
  storeLds(0);

  for (int kt = 0; kt < kTiles; ++kt) {
    if (kt + 1 < kTiles) loadRegs(kt + 1);   // prefetch next tile into regs
    __syncthreads();                          // buf[kt&1] stores visible

    const __bf16* As = AsBase + (kt & 1) * (BM * LDSS);
    const __bf16* Bs = BsBase + (kt & 1) * (BN * LDSS);

    bf16x16 af[4], bfr[2];
#pragma unroll
    for (int mi = 0; mi < 4; ++mi) {
      const __bf16* p = &As[(wm + mi * 16 + m16) * LDSS + kh];
      bf16x8 lo = *(const bf16x8*)p;
      bf16x8 hi = *(const bf16x8*)(p + 16);
      af[mi] = __builtin_shufflevector(lo, hi, 0,1,2,3,4,5,6,7,8,9,10,11,12,13,14,15);
    }
#pragma unroll
    for (int ni = 0; ni < 2; ++ni) {
      const __bf16* p = &Bs[(wn + ni * 16 + m16) * LDSS + kh];
      bf16x8 lo = *(const bf16x8*)p;
      bf16x8 hi = *(const bf16x8*)(p + 16);
      bfr[ni] = __builtin_shufflevector(lo, hi, 0,1,2,3,4,5,6,7,8,9,10,11,12,13,14,15);
    }

#pragma unroll
    for (int mi = 0; mi < 4; ++mi)
#pragma unroll
      for (int ni = 0; ni < 2; ++ni)
        acc[mi][ni] = __builtin_amdgcn_wmma_f32_16x16x32_bf16(
            false, af[mi], false, bfr[ni], (short)0, acc[mi][ni], false, false);

    if (kt + 1 < kTiles) storeLds((kt + 1) & 1);  // other buffer: no sync needed
  }

  __syncthreads();  // all waves done reading As/Bs; reuse smem for C staging

  // Epilogue: bias + ELU, stage into LDS (D layout: N=lane&15, M=(lane>>4)*8+v),
  // then fully-coalesced b128 global stores.
  __bf16* Cs = smem;  // [BM][CSTR]
  const int rowb = (lane >> 4) * 8;
  const int nn   = lane & 15;
#pragma unroll
  for (int mi = 0; mi < 4; ++mi) {
#pragma unroll
    for (int ni = 0; ni < 2; ++ni) {
      const int col = wn + ni * 16 + nn;
      const float bv = bias[n0 + col];
#pragma unroll
      for (int v = 0; v < 8; ++v) {
        const int row = wm + mi * 16 + rowb + v;
        float x = acc[mi][ni][v] + bv;
        if (doElu) x = (x > 0.f) ? x : (__expf(x) - 1.f);
        Cs[row * CSTR + col] = (__bf16)x;
      }
    }
  }
  __syncthreads();

#pragma unroll
  for (int j = 0; j < 8; ++j) {           // 2048 chunks of 16B / 256 threads
    int c   = tid + j * 256;
    int row = c >> 4;
    int cc  = (c & 15) * 8;
    bf16x8 vch = *(const bf16x8*)&Cs[row * CSTR + cc];
    *(bf16x8*)(C + (long)(m0 + row) * ldc + n0 + cc) = vch;
  }
}

// --------------------- fused heads + softmax + combine ----------------------
// One wave per batch row. h3: [E,B,256] bf16, g3: [B,256] bf16,
// w3T: [E,256] bf16, b3: [E] f32, gw3T: [E,256] bf16, gb3: [E] f32.

__global__ void final_combine(const __bf16* __restrict__ h3,
                              const __bf16* __restrict__ g3,
                              const __bf16* __restrict__ w3T,
                              const float*  __restrict__ b3,
                              const __bf16* __restrict__ gw3T,
                              const float*  __restrict__ gb3,
                              float* __restrict__ out, int B) {
  const int gtid = blockIdx.x * blockDim.x + threadIdx.x;
  const int b    = gtid >> 5;
  const int lane = threadIdx.x & 31;
  if (b >= B) return;
  const int k0 = lane * 8;  // 32 lanes x 8 = 256

  float ret[NE], logit[NE];
#pragma unroll
  for (int e = 0; e < NE; ++e) {
    bf16x8 hv = *(const bf16x8*)(h3 + ((long)e * B + b) * 256 + k0);
    bf16x8 wv = *(const bf16x8*)(w3T + e * 256 + k0);
    bf16x8 gv = *(const bf16x8*)(g3 + (long)b * 256 + k0);
    bf16x8 qv = *(const bf16x8*)(gw3T + e * 256 + k0);
    float s = 0.f, t = 0.f;
#pragma unroll
    for (int v = 0; v < 8; ++v) {
      s += (float)hv[v] * (float)wv[v];
      t += (float)gv[v] * (float)qv[v];
    }
#pragma unroll
    for (int off = 16; off > 0; off >>= 1) {
      s += __shfl_xor(s, off, 32);
      t += __shfl_xor(t, off, 32);
    }
    ret[e]   = s + b3[e];
    logit[e] = t + gb3[e];
  }

  float mx = logit[0];
#pragma unroll
  for (int e = 1; e < NE; ++e) mx = fmaxf(mx, logit[e]);
  float den = 0.f, num = 0.f;
#pragma unroll
  for (int e = 0; e < NE; ++e) {
    float w = __expf(logit[e] - mx);
    den += w;
    num += w * ret[e];
  }
  if (lane == 0) out[b] = num / den;
}

// ------------------------------- launcher -----------------------------------

static inline int cdiv(int a, int b) { return (a + b - 1) / b; }

extern "C" void kernel_launch(void* const* d_in, const int* in_sizes, int n_in,
                              void* d_out, int out_size, void* d_ws, size_t ws_size,
                              hipStream_t stream) {
  const float* obs  = (const float*)d_in[0];
  const float* code = (const float*)d_in[1];
  const float* ew0 = (const float*)d_in[2];  const float* eb0 = (const float*)d_in[3];
  const float* ew1 = (const float*)d_in[4];  const float* eb1 = (const float*)d_in[5];
  const float* ew2 = (const float*)d_in[6];  const float* eb2 = (const float*)d_in[7];
  const float* ew3 = (const float*)d_in[8];  const float* eb3 = (const float*)d_in[9];
  const float* gw0 = (const float*)d_in[10]; const float* gb0 = (const float*)d_in[11];
  const float* gw1 = (const float*)d_in[12]; const float* gb1 = (const float*)d_in[13];
  const float* gw2 = (const float*)d_in[14]; const float* gb2 = (const float*)d_in[15];
  const float* gw3 = (const float*)d_in[16]; const float* gb3 = (const float*)d_in[17];
  float* out = (float*)d_out;

  char* ws = (char*)d_ws;
  // ---- workspace layout (bytes) ----
  size_t o = 0;
  __bf16* A0bf  = (__bf16*)(ws + o); o += (size_t)NB * ND * 2;          //  8 MB
  __bf16* Cdbf  = (__bf16*)(ws + o); o += (size_t)NB * NV * 2;          //  1 MB
  __bf16* ew0T  = (__bf16*)(ws + o); o += (size_t)NE * 1024 * 512 * 2;  //  8 MB
  __bf16* ew1T  = (__bf16*)(ws + o); o += (size_t)NE * 512 * 1024 * 2;  //  8 MB
  __bf16* ew2T  = (__bf16*)(ws + o); o += (size_t)NE * 256 * 512 * 2;   //  2 MB
  __bf16* ew3T  = (__bf16*)(ws + o); o += (size_t)NE * 1 * 256 * 2;
  __bf16* gw0T  = (__bf16*)(ws + o); o += (size_t)1024 * 64 * 2;
  __bf16* gw1T  = (__bf16*)(ws + o); o += (size_t)512 * 1024 * 2;
  __bf16* gw2T  = (__bf16*)(ws + o); o += (size_t)256 * 512 * 2;
  __bf16* gw3T  = (__bf16*)(ws + o); o += (size_t)8 * 256 * 2;
  size_t actBase = (o + 255) & ~(size_t)255;
  __bf16* h1 = (__bf16*)(ws + actBase);                                  // [E,B,1024] 128 MB
  __bf16* h2 = (__bf16*)(ws + actBase + (size_t)NE * NB * 1024 * 2);     // [E,B,512]   64 MB
  __bf16* h3 = (__bf16*)(ws + actBase + (size_t)NE * NB * 1024 * 2
                                     + (size_t)NE * NB * 512 * 2);       // [E,B,256]   32 MB
  // gate activations reuse the h1 region (h1 is dead once expert L1 finishes)
  __bf16* g1 = h1;                                                       // [B,1024] 16 MB
  __bf16* g2 = (__bf16*)((char*)g1 + (size_t)NB * 1024 * 2);             // [B,512]   8 MB
  __bf16* g3 = (__bf16*)((char*)g2 + (size_t)NB * 512 * 2);              // [B,256]   4 MB

  // ---- casts + weight transposes ----
  cast_f32_to_bf16<<<cdiv(NB * ND, 256), 256, 0, stream>>>(obs, A0bf, NB * ND);
  cast_f32_to_bf16<<<cdiv(NB * NV, 256), 256, 0, stream>>>(code, Cdbf, NB * NV);
  transpose_cast_bf16<<<cdiv(NE * 512 * 1024, 256), 256, 0, stream>>>(ew0, ew0T, NE, 512, 1024);
  transpose_cast_bf16<<<cdiv(NE * 1024 * 512, 256), 256, 0, stream>>>(ew1, ew1T, NE, 1024, 512);
  transpose_cast_bf16<<<cdiv(NE * 512 * 256, 256), 256, 0, stream>>>(ew2, ew2T, NE, 512, 256);
  transpose_cast_bf16<<<cdiv(NE * 256 * 1, 256), 256, 0, stream>>>(ew3, ew3T, NE, 256, 1);
  transpose_cast_bf16<<<cdiv(64 * 1024, 256), 256, 0, stream>>>(gw0, gw0T, 1, 64, 1024);
  transpose_cast_bf16<<<cdiv(1024 * 512, 256), 256, 0, stream>>>(gw1, gw1T, 1, 1024, 512);
  transpose_cast_bf16<<<cdiv(512 * 256, 256), 256, 0, stream>>>(gw2, gw2T, 1, 512, 256);
  transpose_cast_bf16<<<cdiv(256 * 8, 256), 256, 0, stream>>>(gw3, gw3T, 1, 256, 8);

  // ---- expert MLP chain (batched over E via grid.z) ----
  // L0: [8192x512] x [512x1024] -> h1   (A shared across experts)
  gemm_bf16_wmma<<<dim3(1024 / BN, NB / BM, NE), 256, 0, stream>>>(
      A0bf, 0L, ND, ew0T, (long)1024 * 512, eb0, 1024,
      h1, (long)NB * 1024, 1024, ND, 1);
  // L1: [8192x1024] x [1024x512] -> h2
  gemm_bf16_wmma<<<dim3(512 / BN, NB / BM, NE), 256, 0, stream>>>(
      h1, (long)NB * 1024, 1024, ew1T, (long)512 * 1024, eb1, 512,
      h2, (long)NB * 512, 512, 1024, 1);
  // L2: [8192x512] x [512x256] -> h3
  gemm_bf16_wmma<<<dim3(256 / BN, NB / BM, NE), 256, 0, stream>>>(
      h2, (long)NB * 512, 512, ew2T, (long)256 * 512, eb2, 256,
      h3, (long)NB * 256, 256, 512, 1);

  // ---- gate MLP chain (grid.z = 1) ----
  gemm_bf16_wmma<<<dim3(1024 / BN, NB / BM, 1), 256, 0, stream>>>(
      Cdbf, 0L, NV, gw0T, 0L, gb0, 0, g1, 0L, 1024, NV, 1);
  gemm_bf16_wmma<<<dim3(512 / BN, NB / BM, 1), 256, 0, stream>>>(
      g1, 0L, 1024, gw1T, 0L, gb1, 0, g2, 0L, 512, 1024, 1);
  gemm_bf16_wmma<<<dim3(256 / BN, NB / BM, 1), 256, 0, stream>>>(
      g2, 0L, 512, gw2T, 0L, gb2, 0, g3, 0L, 256, 512, 1);

  // ---- fused heads + softmax + weighted combine ----
  final_combine<<<cdiv(NB * 32, 256), 256, 0, stream>>>(
      h3, g3, ew3T, eb3, gw3T, gb3, out, NB);

  (void)in_sizes; (void)n_in; (void)out_size; (void)ws_size;
}
